// MemoryBlock_70308614636110
// MI455X (gfx1250) — matrix operand order repaired and checked by reading, not compiled, not run on hardware
//
#include <hip/hip_runtime.h>
#include <hip/hip_bf16.h>

// ---- problem constants (match reference) ----
#define BATCH 32
#define LSEQ  1024
#define DMODEL 512
#define NHEAD 8
#define NGRP  65          // G
#define GPAD  80          // G padded to multiple of 16
#define TOPK  16
#define DHEAD 64

typedef __attribute__((ext_vector_type(16))) __bf16 v16bf;
typedef __attribute__((ext_vector_type(8)))  float  v8f;

union FragU { v16bf v; unsigned u[8]; };

__device__ __forceinline__ unsigned short f2bf(float f) {
    union { __bf16 h; unsigned short u; } cv;
    cv.h = (__bf16)f;
    return cv.u;
}

__device__ __forceinline__ unsigned encf(float f) {
    unsigned u = __float_as_uint(f);
    return (u & 0x80000000u) ? ~u : (u | 0x80000000u);
}
__device__ __forceinline__ float decf(unsigned e) {
    unsigned u = (e & 0x80000000u) ? (e & 0x7fffffffu) : ~e;
    return __uint_as_float(u);
}

__device__ __forceinline__ v8f wmma_bf16(v16bf a, v16bf b, v8f c) {
    // D(16x16 f32) = A(16x32 bf16) x B(32x16 bf16) + C
    return __builtin_amdgcn_wmma_f32_16x16x32_bf16(false, a, false, b, (short)0, c, false, false);
}

// A fragment: row m = lane&15 ; lanes<16 hold K klo=0, lanes>=16 klo=8
// VGPR j(0..3): K=klo+2j,+1 ; VGPR j(4..7): K=16+klo+2(j-4),+1   (ISA 7.12.2)
template <bool GUARD>
__device__ __forceinline__ v16bf load_a_frag(const unsigned short* A, int lda,
                                             int m_base, int ks, int lane, int M) {
    int m   = m_base + (lane & 15);
    int klo = (lane < 16) ? 0 : 8;
    FragU f;
    if (!GUARD || m < M) {
        const unsigned short* row = A + (size_t)m * lda + ks;
#pragma unroll
        for (int j = 0; j < 4; ++j) f.u[j]     = *(const unsigned*)(row + klo + 2 * j);
#pragma unroll
        for (int j = 0; j < 4; ++j) f.u[4 + j] = *(const unsigned*)(row + 16 + klo + 2 * j);
    } else {
#pragma unroll
        for (int j = 0; j < 8; ++j) f.u[j] = 0u;
    }
    return f.v;
}

// B fragment from "Bt" layout: Bt[n][k] (K contiguous per column, row stride ldb).
// col n = lane&15 ; lanes<16 hold K 0..7,16..23 of the kstep, lanes>=16 hold K 8..15,24..31.
__device__ __forceinline__ v16bf load_b_frag(const unsigned short* Bt, int ldb,
                                             int n_base, int ks, int lane) {
    int n  = n_base + (lane & 15);
    int kb = (lane < 16) ? 0 : 8;
    const unsigned short* col = Bt + (size_t)n * ldb + ks + kb;
    FragU f;
#pragma unroll
    for (int j = 0; j < 8; ++j) f.u[j] = *(const unsigned*)(col + 2 * (j & 3) + ((j >> 2) << 4));
    return f.v;
}

// ------------------- prep kernels -------------------

__global__ void init_mm_kernel(unsigned* mm) {
    mm[0] = 0xFFFFFFFFu;   // running min (encoded)
    mm[1] = 0u;            // running max (encoded)
}

__global__ void xform_kernel(const float* __restrict__ x, unsigned short* __restrict__ xt, int n) {
    int stride = gridDim.x * blockDim.x;
    for (int i = blockIdx.x * blockDim.x + threadIdx.x; i < n; i += stride)
        xt[i] = f2bf(__logf(fmaxf(x[i], 0.f) + 1.f));
}

// Wt[n*Kd + k] = bf16(W[k*N + n])  (transpose so B-fragment pairs are contiguous)
__global__ void wt_convert_kernel(const float* __restrict__ W, unsigned short* __restrict__ Wt) {
    int i = blockIdx.x * blockDim.x + threadIdx.x;
    if (i >= DMODEL * DMODEL) return;
    int n = i / DMODEL, k = i % DMODEL;
    Wt[(size_t)n * DMODEL + k] = f2bf(W[(size_t)k * DMODEL + n]);
}

// qb[GPAD][D] bf16 = target@Wq + bq, rows >= NGRP zeroed (batch independent)
__global__ void qproj_kernel(const float* __restrict__ target, const float* __restrict__ Wq,
                             const float* __restrict__ bq, unsigned short* __restrict__ qb) {
    int i = blockIdx.x * blockDim.x + threadIdx.x;
    if (i >= GPAD * DMODEL) return;
    int g = i >> 9, n = i & (DMODEL - 1);
    float acc = 0.f;
    if (g < NGRP) {
        acc = bq[n];
        for (int d = 0; d < DMODEL; ++d)
            acc += target[g * DMODEL + d] * Wq[(size_t)d * DMODEL + n];
    }
    qb[i] = f2bf(acc);
}

// ------------------- generic WMMA GEMM -------------------
// Out[M,N] = A[M,Kd](bf16) @ Bt[N,Kd](bf16, K-major) + bias[N]
// block: 256 thr = 8 waves (2 along M x 4 along N); wave tile 64(M)x32(N)
// -> block tile 128x128 ; per K-step per wave: 4 A frags + 2 B frags, 8 WMMAs
template <bool GUARD>
__global__ __launch_bounds__(256) void gemm_bf16_kernel(
        const unsigned short* __restrict__ A, const unsigned short* __restrict__ Bt,
        const float* __restrict__ bias, void* __restrict__ Out,
        int M, int N, int Kd, int out_is_bf16) {
    int lane = threadIdx.x & 31;
    int w    = threadIdx.x >> 5;
    int wm   = w & 1, wn = w >> 1;
    int m0   = blockIdx.y * 128 + wm * 64;
    int n0   = blockIdx.x * 128 + wn * 32;

    v8f acc[4][2];
#pragma unroll
    for (int i = 0; i < 4; ++i)
#pragma unroll
        for (int j = 0; j < 2; ++j) acc[i][j] = (v8f){};

    for (int ks = 0; ks < Kd; ks += 32) {
        v16bf a0 = load_a_frag<GUARD>(A, Kd, m0,      ks, lane, M);
        v16bf a1 = load_a_frag<GUARD>(A, Kd, m0 + 16, ks, lane, M);
        v16bf a2 = load_a_frag<GUARD>(A, Kd, m0 + 32, ks, lane, M);
        v16bf a3 = load_a_frag<GUARD>(A, Kd, m0 + 48, ks, lane, M);
        v16bf b0 = load_b_frag(Bt, Kd, n0,      ks, lane);
        v16bf b1 = load_b_frag(Bt, Kd, n0 + 16, ks, lane);
        acc[0][0] = wmma_bf16(a0, b0, acc[0][0]);
        acc[0][1] = wmma_bf16(a0, b1, acc[0][1]);
        acc[1][0] = wmma_bf16(a1, b0, acc[1][0]);
        acc[1][1] = wmma_bf16(a1, b1, acc[1][1]);
        acc[2][0] = wmma_bf16(a2, b0, acc[2][0]);
        acc[2][1] = wmma_bf16(a2, b1, acc[2][1]);
        acc[3][0] = wmma_bf16(a3, b0, acc[3][0]);
        acc[3][1] = wmma_bf16(a3, b1, acc[3][1]);
    }

    int nn   = lane & 15;
    int moff = (lane < 16) ? 0 : 8;
#pragma unroll
    for (int i = 0; i < 4; ++i) {
#pragma unroll
        for (int j = 0; j < 2; ++j) {
            int mb = m0 + i * 16;
            int n  = n0 + j * 16 + nn;
            float bv = bias[n];
#pragma unroll
            for (int r = 0; r < 8; ++r) {
                int m = mb + moff + r;
                if (!GUARD || m < M) {
                    float v = acc[i][j][r] + bv;
                    if (out_is_bf16) ((unsigned short*)Out)[(size_t)m * N + n] = f2bf(v);
                    else             ((float*)Out)[(size_t)m * N + n] = v;
                }
            }
        }
    }
}

// ------------------- fused scores + top-k + combine -------------------
// grid (gtile=GPAD/16, h=NHEAD, b=BATCH), 1024 threads (32 waves).
// Softmax & the dh^-0.5 scale are strictly monotone per score row and the
// attention values themselves are unused downstream -> top-16 of raw scores
// (lowest index on ties, like jax.lax.top_k) is exactly equivalent.
__global__ __launch_bounds__(1024) void scores_topk_kernel(
        const unsigned short* __restrict__ qb, const unsigned short* __restrict__ kb,
        const float* __restrict__ vf, const float* __restrict__ conv_w,
        const float* __restrict__ conv_b, float* __restrict__ yraw, unsigned* __restrict__ mm) {
    __shared__ float ssc[16 * LSEQ];   // 64 KB: 16 score rows x L (WGP has 320 KB LDS)

    int b = blockIdx.z, h = blockIdx.y, gt = blockIdx.x;
    int lane = threadIdx.x & 31, w = threadIdx.x >> 5;
    int g0 = gt * 16;

    // ---- phase 1: S[16, 1024] via WMMA; wave w owns columns [w*32, w*32+32) ----
    {
        int l0 = w * 32;
        v8f c0 = {}, c1 = {};
        const unsigned short* kbb = kb + (size_t)b * (LSEQ * DMODEL) + h * DHEAD;
        const unsigned short* qbb = qb + h * DHEAD;
#pragma unroll
        for (int ks = 0; ks < DHEAD; ks += 32) {
            v16bf a  = load_a_frag<false>(qbb, DMODEL, g0, ks, lane, GPAD);
            v16bf b0 = load_b_frag(kbb, DMODEL, l0,      ks, lane);
            v16bf b1 = load_b_frag(kbb, DMODEL, l0 + 16, ks, lane);
            c0 = wmma_bf16(a, b0, c0);
            c1 = wmma_bf16(a, b1, c1);
        }
        int nn = lane & 15, moff = (lane < 16) ? 0 : 8;
#pragma unroll
        for (int r = 0; r < 8; ++r) {
            ssc[(moff + r) * LSEQ + l0 + nn]      = c0[r];
            ssc[(moff + r) * LSEQ + l0 + 16 + nn] = c1[r];
        }
    }
    __syncthreads();

    // ---- phase 2: wave w (<16) handles score row g = g0 + w ----
    if (w < 16) {
        int g = g0 + w;
        if (g < NGRP) {
            const float NEG = -__builtin_inff();
            float rv[32];               // register-resident row slice: l = t*32 + lane
#pragma unroll
            for (int t = 0; t < 32; ++t) rv[t] = ssc[w * LSEQ + t * 32 + lane];

            int kidx[TOPK];
#pragma unroll
            for (int it = 0; it < TOPK; ++it) {
                float bv = NEG; int bi = 0x7fffffff;
#pragma unroll
                for (int t = 0; t < 32; ++t) {
                    int l = t * 32 + lane;
                    if (rv[t] > bv) { bv = rv[t]; bi = l; }   // strict > keeps lowest l
                }
#pragma unroll
                for (int off = 16; off >= 1; off >>= 1) {
                    float ov = __shfl_xor(bv, off, 32);
                    int   oi = __shfl_xor(bi, off, 32);
                    if (ov > bv || (ov == bv && oi < bi)) { bv = ov; bi = oi; }
                }
                kidx[it] = bi;
                int bt = bi >> 5, bl = bi & 31;
#pragma unroll
                for (int t = 0; t < 32; ++t)
                    if (t == bt && lane == bl) rv[t] = NEG;   // knock out winner
            }

            // ---- combine: y[b,h,g,:] = sum_r conv_w[g,r]*v[b,h,idx_r,:] + conv_b[g] ----
            float cb = conv_b[g];
            float a0 = cb, a1 = cb;     // channels d=lane and d=lane+32
#pragma unroll
            for (int r = 0; r < TOPK; ++r) {
                float wv = conv_w[g * TOPK + r];
                const float* vrow = vf + ((size_t)(b * LSEQ + kidx[r])) * DMODEL + h * DHEAD;
                a0 += wv * vrow[lane];
                a1 += wv * vrow[lane + 32];
            }
            size_t yo = ((size_t)(b * NGRP + g)) * DMODEL + h * DHEAD;  // [b,g,(h,d)] layout
            yraw[yo + lane]      = a0;
            yraw[yo + 32 + lane] = a1;

            float mn = fminf(a0, a1), mx = fmaxf(a0, a1);
#pragma unroll
            for (int off = 16; off >= 1; off >>= 1) {
                mn = fminf(mn, __shfl_xor(mn, off, 32));
                mx = fmaxf(mx, __shfl_xor(mx, off, 32));
            }
            if (lane == 0) {
                atomicMin(&mm[0], encf(mn));
                atomicMax(&mm[1], encf(mx));
            }
        }
    }
}

// ------------------- finalize + normalize -------------------

__global__ void finalize_mm_kernel(const unsigned* __restrict__ mm, float* __restrict__ params) {
    float mn = decf(mm[0]), mx = decf(mm[1]);
    float d = mx - mn;
    params[0] = mn;
    params[1] = (d != 0.f) ? (1.f / d) : 1.f;
}

__global__ void normalize_kernel(const float* __restrict__ yraw, const float* __restrict__ params,
                                 unsigned short* __restrict__ ybf, int n) {
    float mn = params[0], inv = params[1];
    int stride = gridDim.x * blockDim.x;
    for (int i = blockIdx.x * blockDim.x + threadIdx.x; i < n; i += stride)
        ybf[i] = f2bf(__expf((yraw[i] - mn) * inv));
}

// ------------------- driver -------------------

extern "C" void kernel_launch(void* const* d_in, const int* in_sizes, int n_in,
                              void* d_out, int out_size, void* d_ws, size_t ws_size,
                              hipStream_t stream) {
    const float* x      = (const float*)d_in[0];
    const float* Wq     = (const float*)d_in[1];
    const float* bq     = (const float*)d_in[2];
    const float* Wk     = (const float*)d_in[3];
    const float* bk     = (const float*)d_in[4];
    const float* Wv     = (const float*)d_in[5];
    const float* bv     = (const float*)d_in[6];
    const float* Wo     = (const float*)d_in[7];
    const float* bo     = (const float*)d_in[8];
    const float* target = (const float*)d_in[9];
    const float* conv_w = (const float*)d_in[10];
    const float* conv_b = (const float*)d_in[11];
    float* out = (float*)d_out;

    // workspace carve-up (256B aligned)
    char* ws = (char*)d_ws;
    size_t off = 0;
    auto alloc = [&](size_t bytes) { void* p = ws + off; off = (off + bytes + 255) & ~(size_t)255; return p; };
    const size_t nXT  = (size_t)BATCH * LSEQ * DMODEL;
    const size_t MOUT = (size_t)BATCH * NGRP;           // 2080
    unsigned short* xt   = (unsigned short*)alloc(nXT * 2);
    unsigned short* kbuf = (unsigned short*)alloc(nXT * 2);
    float*          vf   = (float*)alloc(nXT * 4);
    unsigned short* wkt  = (unsigned short*)alloc((size_t)DMODEL * DMODEL * 2);
    unsigned short* wvt  = (unsigned short*)alloc((size_t)DMODEL * DMODEL * 2);
    unsigned short* wot  = (unsigned short*)alloc((size_t)DMODEL * DMODEL * 2);
    unsigned short* qb   = (unsigned short*)alloc((size_t)GPAD * DMODEL * 2);
    float*          yraw = (float*)alloc(MOUT * DMODEL * 4);
    unsigned short* ybf  = (unsigned short*)alloc(MOUT * DMODEL * 2);
    unsigned*       mm   = (unsigned*)alloc(64);
    float*          prm  = (float*)alloc(64);
    (void)ws_size; (void)in_sizes; (void)n_in; (void)out_size;

    init_mm_kernel<<<1, 1, 0, stream>>>(mm);
    xform_kernel<<<4096, 256, 0, stream>>>(x, xt, (int)nXT);
    wt_convert_kernel<<<(DMODEL * DMODEL + 255) / 256, 256, 0, stream>>>(Wk, wkt);
    wt_convert_kernel<<<(DMODEL * DMODEL + 255) / 256, 256, 0, stream>>>(Wv, wvt);
    wt_convert_kernel<<<(DMODEL * DMODEL + 255) / 256, 256, 0, stream>>>(Wo, wot);
    qproj_kernel<<<(GPAD * DMODEL + 255) / 256, 256, 0, stream>>>(target, Wq, bq, qb);

    // K and V projections: M=32768, N=512, Kd=512 (exact 128x128 block tiles -> no guard)
    {
        dim3 grid(DMODEL / 128, (BATCH * LSEQ) / 128);
        gemm_bf16_kernel<false><<<grid, 256, 0, stream>>>(xt, wkt, bk, kbuf, BATCH * LSEQ, DMODEL, DMODEL, 1);
        gemm_bf16_kernel<false><<<grid, 256, 0, stream>>>(xt, wvt, bv, vf,   BATCH * LSEQ, DMODEL, DMODEL, 0);
    }

    // fused scores + top-k + gather/combine
    {
        dim3 grid(GPAD / 16, NHEAD, BATCH);
        scores_topk_kernel<<<grid, 1024, 0, stream>>>(qb, kbuf, vf, conv_w, conv_b, yraw, mm);
    }

    finalize_mm_kernel<<<1, 1, 0, stream>>>(mm, prm);
    normalize_kernel<<<2080, 256, 0, stream>>>(yraw, prm, ybf, (int)(MOUT * DMODEL));

    // out[M=2080, 512] = ybf @ Wo + bo (ragged last block row -> guarded version)
    {
        dim3 grid(DMODEL / 128, (unsigned)((MOUT + 127) / 128));
        gemm_bf16_kernel<true><<<grid, 256, 0, stream>>>(ybf, wot, bo, out, (int)MOUT, DMODEL, DMODEL, 0);
    }
}
